// _Joiner_67302137528860
// MI455X (gfx1250) — matrix operand order repaired and checked by reading, not compiled, not run on hardware
//
#include <hip/hip_runtime.h>
#include <stdint.h>

typedef _Float16 half_t;
typedef __attribute__((ext_vector_type(16))) _Float16 v16h;
typedef __attribute__((ext_vector_type(8)))  _Float16 v8h;
typedef __attribute__((ext_vector_type(4)))  _Float16 v4h;
typedef __attribute__((ext_vector_type(8)))  float    v8f;

#define NB     8
#define NT_SEQ 128   // T
#define NU     64    // U
#define ND     512   // D (GEMM K)
#define NV     1024  // V (GEMM N)

#define TILE_M 128
#define TILE_N 128
#define TILE_K 32
#define LDSS   40    // padded LDS row stride in halves (80B, 16B-aligned, conflict-free)

union Q16  { uint4 q;    v8h  h; };
union F16x16 { uint4 q[2]; v16h v; };

__device__ __forceinline__ half_t relu_h(float a, float b) {
    return (half_t)fmaxf(a + b, 0.0f);
}

// ---- pre-pass: convert W (V x D, f32) to f16 in workspace ----
__global__ __launch_bounds__(256) void convert_w_f16(const float* __restrict__ W,
                                                     half_t* __restrict__ Wh) {
    int i = (blockIdx.x * 256 + threadIdx.x) * 4;
    float4 w = *(const float4*)(W + i);
    v4h h;
    h[0] = (half_t)w.x; h[1] = (half_t)w.y; h[2] = (half_t)w.z; h[3] = (half_t)w.w;
    *(v4h*)(Wh + i) = h;
}

// ---- main: fused (broadcast-add + relu) A-generation + WMMA GEMM + bias ----
__global__ __launch_bounds__(256) void joiner_wmma_kernel(
    const float* __restrict__ src,     // (B, T, D)
    const float* __restrict__ tgt,     // (B, U, D)
    const half_t* __restrict__ Wh,     // (V, D) f16
    const float* __restrict__ bias,    // (V)
    float* __restrict__ out)           // (B, T, U, V)
{
    __shared__ half_t lA[TILE_M * LDSS];
    __shared__ half_t lB[TILE_N * LDSS];

    const int n0 = blockIdx.x * TILE_N;
    const int m0 = blockIdx.y * TILE_M;   // row within batch (t*U + u)
    const int bi = blockIdx.z;

    const int tid  = threadIdx.x;
    const int lane = tid & 31;
    const int wid  = tid >> 5;            // 0..7 (8 wave32s)
    const int wm   = (wid & 1) * 64;      // wave M offset inside tile
    const int wn   = (wid >> 1) * 32;     // wave N offset inside tile

    // cooperative staging assignment: 128 rows x 2 half-K chunks
    const int lrow = tid & 127;
    const int lkc  = (tid >> 7) * 16;     // 0 or 16

    const int grow = m0 + lrow;           // 0..8191
    const int tIdx = grow >> 6;           // / U
    const int uIdx = grow & 63;           // % U

    const float*  srcRow = src + ((size_t)bi * NT_SEQ + tIdx) * ND + lkc;
    const float*  tgtRow = tgt + ((size_t)bi * NU + uIdx) * ND + lkc;
    const half_t* wRow   = Wh  + (size_t)(n0 + lrow) * ND + lkc;

    half_t* aStore = &lA[lrow * LDSS + lkc];
    half_t* bStore = &lB[lrow * LDSS + lkc];

    // WMMA fragment addressing (per ISA VGPR layout tables)
    const int mrow  = lane & 15;
    const int koffA = (lane >> 4) * 8;    // A: lanes 0-15 -> K {0..7,16..23}; 16-31 -> {8..15,24..31}
    const int koffB = (lane >> 4) * 8;    // B: lanes 0-15 -> K {0..7,16..23}; 16-31 -> {8..15,24..31}

    const half_t* aFragBase = &lA[(wm + mrow) * LDSS + koffA];
    const half_t* bFragBase = &lB[(wn + mrow) * LDSS + koffB];

    v8f acc[4][2] = {};  // 4 M-subtiles x 2 N-subtiles of 16x16 f32

    for (int k0 = 0; k0 < ND; k0 += TILE_K) {
        // ---- generate A chunk in registers: relu(src + tgt), cvt f16 ----
        float4 s0 = *(const float4*)(srcRow + k0 + 0);
        float4 s1 = *(const float4*)(srcRow + k0 + 4);
        float4 s2 = *(const float4*)(srcRow + k0 + 8);
        float4 s3 = *(const float4*)(srcRow + k0 + 12);
        float4 g0 = *(const float4*)(tgtRow + k0 + 0);
        float4 g1 = *(const float4*)(tgtRow + k0 + 4);
        float4 g2 = *(const float4*)(tgtRow + k0 + 8);
        float4 g3 = *(const float4*)(tgtRow + k0 + 12);

        Q16 p0, p1;
        p0.h[0] = relu_h(s0.x, g0.x); p0.h[1] = relu_h(s0.y, g0.y);
        p0.h[2] = relu_h(s0.z, g0.z); p0.h[3] = relu_h(s0.w, g0.w);
        p0.h[4] = relu_h(s1.x, g1.x); p0.h[5] = relu_h(s1.y, g1.y);
        p0.h[6] = relu_h(s1.z, g1.z); p0.h[7] = relu_h(s1.w, g1.w);
        p1.h[0] = relu_h(s2.x, g2.x); p1.h[1] = relu_h(s2.y, g2.y);
        p1.h[2] = relu_h(s2.z, g2.z); p1.h[3] = relu_h(s2.w, g2.w);
        p1.h[4] = relu_h(s3.x, g3.x); p1.h[5] = relu_h(s3.y, g3.y);
        p1.h[6] = relu_h(s3.z, g3.z); p1.h[7] = relu_h(s3.w, g3.w);

        // ---- B chunk: already f16 ----
        uint4 wb0 = *(const uint4*)(wRow + k0);
        uint4 wb1 = *(const uint4*)(wRow + k0 + 8);

        __syncthreads();  // WAR: prior iteration's fragment reads done
        *(uint4*)(aStore)     = p0.q;
        *(uint4*)(aStore + 8) = p1.q;
        *(uint4*)(bStore)     = wb0;
        *(uint4*)(bStore + 8) = wb1;
        __syncthreads();  // RAW: tiles visible

        // prefetch next K chunk (global_prefetch_b8)
        if (k0 + TILE_K < ND) {
            __builtin_prefetch(srcRow + k0 + TILE_K, 0, 0);
            __builtin_prefetch(tgtRow + k0 + TILE_K, 0, 0);
            __builtin_prefetch(wRow   + k0 + TILE_K, 0, 0);
        }

        // ---- load fragments from LDS ----
        F16x16 bf[2];
        #pragma unroll
        for (int nt = 0; nt < 2; ++nt) {
            const half_t* p = bFragBase + nt * 16 * LDSS;
            bf[nt].q[0] = *(const uint4*)(p);       // K koffB..koffB+7
            bf[nt].q[1] = *(const uint4*)(p + 16);  // K koffB+16..+23
        }
        F16x16 af[4];
        #pragma unroll
        for (int mt = 0; mt < 4; ++mt) {
            const half_t* p = aFragBase + mt * 16 * LDSS;
            af[mt].q[0] = *(const uint4*)(p);       // K koffA..koffA+7
            af[mt].q[1] = *(const uint4*)(p + 16);  // K koffA+16..+23
        }

        // ---- 8 WMMAs ----
        #pragma unroll
        for (int mt = 0; mt < 4; ++mt)
            #pragma unroll
            for (int nt = 0; nt < 2; ++nt)
                acc[mt][nt] = __builtin_amdgcn_wmma_f32_16x16x32_f16(
                    false, af[mt].v, false, bf[nt].v,
                    (short)0, acc[mt][nt], false, false);
    }

    // ---- epilogue: bias add + store (C/D layout: M = r + 8*(lane>=16), N = lane%16) ----
    const int moff = (lane >> 4) * 8;
    #pragma unroll
    for (int nt = 0; nt < 2; ++nt) {
        const int col = n0 + wn + nt * 16 + mrow;
        const float bb = bias[col];
        #pragma unroll
        for (int mt = 0; mt < 4; ++mt) {
            const size_t rowBase = (size_t)bi * (NT_SEQ * NU) + (size_t)(m0 + wm + mt * 16 + moff);
            float* op = out + rowBase * NV + col;
            #pragma unroll
            for (int r = 0; r < 8; ++r)
                op[(size_t)r * NV] = acc[mt][nt][r] + bb;
        }
    }
}

extern "C" void kernel_launch(void* const* d_in, const int* in_sizes, int n_in,
                              void* d_out, int out_size, void* d_ws, size_t ws_size,
                              hipStream_t stream) {
    const float* src  = (const float*)d_in[0];
    const int*   slen = (const int*)d_in[1];
    const float* tgt  = (const float*)d_in[2];
    const int*   tlen = (const int*)d_in[3];
    const float* W    = (const float*)d_in[4];
    const float* bias = (const float*)d_in[5];
    float*  out = (float*)d_out;
    half_t* Wh  = (half_t*)d_ws;   // V*D f16 = 1 MB scratch

    // 1) W -> f16 (524288 elems, 4 per thread)
    convert_w_f16<<<(NV * ND) / (256 * 4), 256, 0, stream>>>(W, Wh);

    // 2) fused joiner GEMM
    dim3 grid(NV / TILE_N, (NT_SEQ * NU) / TILE_M, NB);
    joiner_wmma_kernel<<<grid, 256, 0, stream>>>(src, tgt, Wh, bias, out);

    // 3) tuple pass-through: lengths bit-copied after the main output
    const size_t mainElems = (size_t)NB * NT_SEQ * NU * NV;
    hipMemcpyAsync(out + mainElems,      slen, NB * sizeof(int),
                   hipMemcpyDeviceToDevice, stream);
    hipMemcpyAsync(out + mainElems + NB, tlen, NB * sizeof(int),
                   hipMemcpyDeviceToDevice, stream);
}
